// HierarchicalAttention_54039278518787
// MI455X (gfx1250) — matrix operand order repaired and checked by reading, not compiled, not run on hardware
//
#include <hip/hip_runtime.h>
#include <hip/hip_bf16.h>

// Problem constants (from reference): B=32, N=8192, D=128, C=128, HID=64
#define B_ 32
#define N_ 8192
#define D_ 128
#define C_ 128
#define HID_ 64

typedef __attribute__((ext_vector_type(16))) __bf16 v16bf;
typedef __attribute__((ext_vector_type(8)))  float  v8f;

// float -> bf16 with round-to-nearest-even (bit-level, avoids relying on __bf16 casts)
__device__ __forceinline__ unsigned short f2bf(float f) {
    unsigned int u = __float_as_uint(f);
    unsigned int r = u + 0x7FFFu + ((u >> 16) & 1u);
    return (unsigned short)(r >> 16);
}

union AFrag { v16bf v; unsigned short u[16]; };

// ---------------------------------------------------------------------------
// zero-init kernel (out + denom are atomically accumulated, must start at 0)
// ---------------------------------------------------------------------------
__global__ void zero_kernel(float* p, int n) {
    int i = blockIdx.x * blockDim.x + threadIdx.x;
    int stride = gridDim.x * blockDim.x;
    for (; i < n; i += stride) p[i] = 0.0f;
}

// ---------------------------------------------------------------------------
// Kernel 1: importance MLP via bf16 WMMA.
//   e[g]      = exp(sigmoid(relu(F[g]·w1 + b1)·w2 + b2))
//   denom[b,c] += e[g]   (global fp32 atomic)
// Each wave handles 16 agents; block = 256 threads = 8 waves = 128 agents.
// ---------------------------------------------------------------------------
__global__ void imp_kernel(const float* __restrict__ feat,
                           const float* __restrict__ w1,
                           const float* __restrict__ b1,
                           const float* __restrict__ w2,
                           const float* __restrict__ b2,
                           const int*   __restrict__ clus,
                           float* __restrict__ e_out,
                           float* __restrict__ denom) {
    // w1 as bf16, pre-swizzled into WMMA B-fragment order:
    // frag[ktile][ntile][lane][e]; lane<16 holds K=ktile*32+e, lane>=16 holds K=ktile*32+16+e,
    // column n = ntile*16 + (lane&15). 4*4*32*16*2B = 16 KB.
    __shared__ __align__(32) unsigned short sw1[4][4][32][16];

    const int tid  = threadIdx.x;
    const int lane = tid & 31;
    const int wv   = tid >> 5;

    // ---- build w1 bf16 fragments in LDS (once per block) ----
    for (int idx = tid; idx < 4 * 4 * 32; idx += blockDim.x) {
        const int ktile = idx >> 7;          // /128
        const int ntile = (idx >> 5) & 3;    // /32 %4
        const int ln    = idx & 31;
        const int kbase = ktile * 32 + ((ln < 16) ? 0 : 16);
        const int n     = ntile * 16 + (ln & 15);
#pragma unroll
        for (int e = 0; e < 16; ++e)
            sw1[ktile][ntile][ln][e] = f2bf(w1[(kbase + e) * HID_ + n]);
    }
    __syncthreads();

    // ---- 16 agents per wave ----
    const long tileBase = (long)blockIdx.x * 128 + wv * 16;   // first agent of this wave's tile
    const int  aRow     = (int)(lane & 15);                   // A-matrix row M for this lane
    const long fbase    = (tileBase + aRow) * (long)D_;

    // Load A fragments for all 4 K-tiles (lane<16: K = kt*32+{0..15}; lane>=16: +16)
    AFrag afr[4];
#pragma unroll
    for (int kt = 0; kt < 4; ++kt) {
        const int kb = kt * 32 + ((lane < 16) ? 0 : 16);
        const float4 c0 = *reinterpret_cast<const float4*>(feat + fbase + kb);
        const float4 c1 = *reinterpret_cast<const float4*>(feat + fbase + kb + 4);
        const float4 c2 = *reinterpret_cast<const float4*>(feat + fbase + kb + 8);
        const float4 c3 = *reinterpret_cast<const float4*>(feat + fbase + kb + 12);
        afr[kt].u[0]  = f2bf(c0.x); afr[kt].u[1]  = f2bf(c0.y);
        afr[kt].u[2]  = f2bf(c0.z); afr[kt].u[3]  = f2bf(c0.w);
        afr[kt].u[4]  = f2bf(c1.x); afr[kt].u[5]  = f2bf(c1.y);
        afr[kt].u[6]  = f2bf(c1.z); afr[kt].u[7]  = f2bf(c1.w);
        afr[kt].u[8]  = f2bf(c2.x); afr[kt].u[9]  = f2bf(c2.y);
        afr[kt].u[10] = f2bf(c2.z); afr[kt].u[11] = f2bf(c2.w);
        afr[kt].u[12] = f2bf(c3.x); afr[kt].u[13] = f2bf(c3.y);
        afr[kt].u[14] = f2bf(c3.z); afr[kt].u[15] = f2bf(c3.w);
    }

    // h = F·w1 : 4 N-tiles x 4 K-steps of v_wmma_f32_16x16x32_bf16
    float pacc[8];
#pragma unroll
    for (int r = 0; r < 8; ++r) pacc[r] = 0.0f;

#pragma unroll
    for (int nt = 0; nt < 4; ++nt) {
        v8f acc = {};
#pragma unroll
        for (int kt = 0; kt < 4; ++kt) {
            const v16bf bfrag = *reinterpret_cast<const v16bf*>(&sw1[kt][nt][lane][0]);
            acc = __builtin_amdgcn_wmma_f32_16x16x32_bf16(
                false, afr[kt].v, false, bfrag, (short)0, acc, false, false);
        }
        // acc element r: row M = (lane<16 ? r : 8+r), col n = nt*16 + (lane&15)
        const int   n   = nt * 16 + (lane & 15);
        const float bb  = b1[n];
        const float w2n = w2[n];
#pragma unroll
        for (int r = 0; r < 8; ++r) {
            float h = acc[r] + bb;
            h = (h > 0.0f) ? h : 0.0f;     // ReLU
            pacc[r] += h * w2n;            // fold w2 dot into the lane partials
        }
    }

    // reduce over the 16 lanes of each half (sum over hidden units)
#pragma unroll
    for (int r = 0; r < 8; ++r) {
        float v = pacc[r];
        v += __shfl_xor(v, 8, 16);
        v += __shfl_xor(v, 4, 16);
        v += __shfl_xor(v, 2, 16);
        v += __shfl_xor(v, 1, 16);
        pacc[r] = v;
    }

    const float bias2 = b2[0];
    // lanes r (low half) emit agent M=r; lanes 16+r emit agent M=8+r
#pragma unroll
    for (int r = 0; r < 8; ++r) {
        if ((lane & 15) == r) {
            const int  M   = r + ((lane < 16) ? 0 : 8);
            const long g   = tileBase + M;
            const float imp = pacc[r] + bias2;
            const float sg  = 1.0f / (1.0f + expf(-imp));   // sigmoid
            const float e   = expf(sg);                      // max-shift cancels analytically
            e_out[g] = e;
            const int b = (int)(g >> 13);                    // g / N_
            const int c = clus[g];
            unsafeAtomicAdd(&denom[(b << 7) + c], e);        // global_atomic_add_f32
        }
    }
}

// ---------------------------------------------------------------------------
// Kernel 2: softmax-weighted scatter. One block per (batch, N/8 chunk).
// LDS acc[C][D] (64 KB) accumulated with ds_add_f32, flushed with global
// fp32 atomics. Second features read is expected to hit the 192 MB L2.
// ---------------------------------------------------------------------------
#define SPLIT_ 8
#define AGENTS_PER_BLOCK_ (N_ / SPLIT_)   // 1024

__global__ void scatter_kernel(const float* __restrict__ feat,
                               const int*   __restrict__ clus,
                               const float* __restrict__ e_in,
                               const float* __restrict__ denom,
                               float* __restrict__ out) {
    extern __shared__ float acc[];        // C_*D_ = 16384 floats = 64 KB

    const int tid   = threadIdx.x;
    const int lane  = tid & 31;
    const int wv    = tid >> 5;           // 8 waves
    const int b     = blockIdx.x / SPLIT_;
    const int chunk = blockIdx.x % SPLIT_;
    const long gBase = (long)b * N_ + (long)chunk * AGENTS_PER_BLOCK_;

    for (int i = tid; i < C_ * D_; i += blockDim.x) acc[i] = 0.0f;
    __syncthreads();

    // one wave per agent; lane loads float4 of the 128-wide feature row
    for (int i = wv; i < AGENTS_PER_BLOCK_; i += 8) {
        const long g  = gBase + i;
        const int  c  = clus[g];
        const float w = e_in[g] / denom[(b << 7) + c];
        const float4 f = *reinterpret_cast<const float4*>(feat + g * (long)D_ + lane * 4);
        __builtin_prefetch(feat + (g + 8) * (long)D_ + lane * 4, 0, 1);
        float* dst = &acc[c * D_ + lane * 4];
        unsafeAtomicAdd(dst + 0, f.x * w);    // ds_add_f32
        unsafeAtomicAdd(dst + 1, f.y * w);
        unsafeAtomicAdd(dst + 2, f.z * w);
        unsafeAtomicAdd(dst + 3, f.w * w);
    }
    __syncthreads();

    float* outB = out + (long)b * C_ * D_;
    for (int i = tid; i < C_ * D_; i += blockDim.x)
        unsafeAtomicAdd(&outB[i], acc[i]);    // merge the 8 chunks per batch
}

// ---------------------------------------------------------------------------
extern "C" void kernel_launch(void* const* d_in, const int* in_sizes, int n_in,
                              void* d_out, int out_size, void* d_ws, size_t ws_size,
                              hipStream_t stream) {
    const float* feat = (const float*)d_in[0];
    const float* w1   = (const float*)d_in[1];
    const float* b1   = (const float*)d_in[2];
    const float* w2   = (const float*)d_in[3];
    const float* b2   = (const float*)d_in[4];
    const int*   clus = (const int*)d_in[5];
    float* out = (float*)d_out;

    float* e_buf = (float*)d_ws;                       // B*N floats  (1 MB)
    float* denom = e_buf + (size_t)B_ * N_;            // B*C floats (16 KB)

    zero_kernel<<<256, 256, 0, stream>>>(out, B_ * C_ * D_);
    zero_kernel<<<16, 256, 0, stream>>>(denom, B_ * C_);

    // B*N / 128 agents-per-block = 2048 blocks
    imp_kernel<<<(B_ * N_) / 128, 256, 0, stream>>>(feat, w1, b1, w2, b2, clus,
                                                    e_buf, denom);

    scatter_kernel<<<B_ * SPLIT_, 256, C_ * D_ * sizeof(float), stream>>>(
        feat, clus, e_buf, denom, out);
}
